// QuantumFeedForward_65481071405009
// MI455X (gfx1250) — matrix operand order repaired and checked by reading, not compiled, not run on hardware
//
#include <hip/hip_runtime.h>

// ---------------------------------------------------------------------------
// QuantumFeedForward on MI455X (gfx1250, wave32, WMMA)
//
// out[T,512] = relu( proj[T,4] @ W1^T[4,2048] ) @ W2^T[2048,512]
// proj = analytic 4-qubit sim (exact f32, per token).
//
// GEMM2 (137 GFLOP) runs on v_wmma_f32_16x16x32_f16 with f32 accumulation.
// Each block owns 32 tokens: h stripe (32 x 2048 f16 = 128 KB) lives in LDS
// in WMMA-A chunk layout; 16 waves sweep all 512 output columns against it.
// B fragments (W2 in f16, 2 MB, L2-resident) and A fragments are prefetched
// one K-step ahead (explicit double buffering) so WMMAs never wait on the
// loads issued in the same step.
// ---------------------------------------------------------------------------

typedef __attribute__((ext_vector_type(16))) _Float16 v16h;
typedef __attribute__((ext_vector_type(8)))  _Float16 v8h;
typedef __attribute__((ext_vector_type(4)))  _Float16 v4h;
typedef __attribute__((ext_vector_type(8)))  float    v8f;

#define E_DIM   512
#define FF_DIM  2048
#define N_TOK   65536        // 8 * 8192
#define TOKS_PER_BLOCK 32
#define THREADS 512          // 16 waves

__device__ __forceinline__ v16h frag16(v8h lo, v8h hi) {
    return __builtin_shufflevector(lo, hi,
        0,1,2,3,4,5,6,7,8,9,10,11,12,13,14,15);
}

__device__ __forceinline__ v8f wmma_f16(v16h a, v16h b, v8f c) {
    return __builtin_amdgcn_wmma_f32_16x16x32_f16(
        false, a, false, b, (short)0, c, false, false);
}

// ---------------------------------------------------------------------------
// Kernel 1: bake the 8 Rot(phi,theta,omega) 2x2 complex matrices into ws.
// gates[l*4+q][8] = {re00,im00,re01,im01,re10,im10,re11,im11}
// ---------------------------------------------------------------------------
__global__ void qff_prep_gates(const float* __restrict__ params,
                               float* __restrict__ gates) {
    int t = threadIdx.x;          // t = l*4 + q, 8 gates total
    if (t < 8) {
        float phi   = params[t * 3 + 0];
        float theta = params[t * 3 + 1];
        float omega = params[t * 3 + 2];
        float ct = cosf(0.5f * theta);
        float st = sinf(0.5f * theta);
        float pr, pi, mr, mi;
        sincosf(-0.5f * (phi + omega), &pi, &pr);   // ep = exp(-i(phi+omega)/2)
        sincosf(-0.5f * (phi - omega), &mi, &mr);   // em = exp(-i(phi-omega)/2)
        float* g = gates + t * 8;
        g[0] =  pr * ct;  g[1] =  pi * ct;          // U00 = ep*c
        g[2] = -mr * st;  g[3] =  mi * st;          // U01 = -conj(em)*s
        g[4] =  mr * st;  g[5] =  mi * st;          // U10 = em*s
        g[6] =  pr * ct;  g[7] = -pi * ct;          // U11 = conj(ep)*c
    }
}

// ---------------------------------------------------------------------------
// Kernel 2: W2 [512,2048] f32 -> f16 (same row-major layout). 2 MB, L2-hot.
// ---------------------------------------------------------------------------
__global__ void qff_cvt_w2(const float* __restrict__ W2, v4h* __restrict__ W2h) {
    int i = blockIdx.x * blockDim.x + threadIdx.x;   // 262144 float4 groups
    float4 v = ((const float4*)W2)[i];
    v4h h;
    h[0] = (_Float16)v.x; h[1] = (_Float16)v.y;
    h[2] = (_Float16)v.z; h[3] = (_Float16)v.w;
    W2h[i] = h;
}

// ---------------------------------------------------------------------------
// Kernel 3: fused sim + GEMM1(relu) + GEMM2.
// Block = 512 threads (16 waves), owns 32 tokens (2 M-tiles).
// LDS: h stripe as 8-half chunks: Hc[kchunk(0..255)][row(0..31)], 128 KB.
// Wave w handles N-tiles {w, w+16} x M-tiles {0,1}: 4 accumulators,
// 4 wmma per K-step, fragments double-buffered one K-step ahead.
// ---------------------------------------------------------------------------
__global__ __launch_bounds__(THREADS) void qff_main(
        const float* __restrict__ x,        // [T, 512] rows; angles = first 4
        const float* __restrict__ W1,       // [2048, 4] row-major
        const _Float16* __restrict__ W2h,   // [512, 2048] f16 row-major
        const float* __restrict__ gates,    // [8][8]
        float* __restrict__ out)            // [T, 512] f32
{
    extern __shared__ __align__(16) unsigned char smem[];
    v8h* __restrict__ Hc = (v8h*)smem;      // [256 kchunks][32 rows], 16 B each

    const int tid = threadIdx.x;
    const int m   = tid & 31;                       // token row within block
    const int tok = blockIdx.x * TOKS_PER_BLOCK + m;

    // ---- quantum sim (exact f32) for token `tok` (16x redundant, ~1k flop) --
    float4 ang = *(const float4*)(x + (size_t)tok * E_DIM);
    float av[4] = {ang.x, ang.y, ang.z, ang.w};
    float cq[4], sq[4];
#pragma unroll
    for (int q = 0; q < 4; ++q) sincosf(0.5f * av[q], &sq[q], &cq[q]);

    // state[idx], idx = b0*8 + b1*4 + b2*2 + b3 (b0 = qubit 0)
    float sr[16], si[16];
#pragma unroll
    for (int idx = 0; idx < 16; ++idx) {
        float pr = 1.f, pi = 0.f;
#pragma unroll
        for (int q = 0; q < 4; ++q) {
            int bit = (idx >> (3 - q)) & 1;
            // amp0 = (c, 0); amp1 = (0, -s)   [RX(x)|0>]
            float br = bit ? 0.f : cq[q];
            float bi = bit ? -sq[q] : 0.f;
            float nr = pr * br - pi * bi;
            float ni = pr * bi + pi * br;
            pr = nr; pi = ni;
        }
        sr[idx] = pr; si[idx] = pi;
    }

#pragma unroll
    for (int l = 0; l < 2; ++l) {
#pragma unroll
        for (int q = 0; q < 4; ++q) {
            const float* g = gates + ((l << 2) + q) * 8;
            float u00r = g[0], u00i = g[1], u01r = g[2], u01i = g[3];
            float u10r = g[4], u10i = g[5], u11r = g[6], u11i = g[7];
            const int stride = 8 >> q;              // 1 << (3-q)
#pragma unroll
            for (int i0 = 0; i0 < 16; ++i0) {
                if (i0 & stride) continue;          // resolved at compile time
                int i1 = i0 + stride;
                float r0 = sr[i0], m0 = si[i0], r1 = sr[i1], m1 = si[i1];
                sr[i0] = u00r * r0 - u00i * m0 + u01r * r1 - u01i * m1;
                si[i0] = u00r * m0 + u00i * r0 + u01r * m1 + u01i * r1;
                sr[i1] = u10r * r0 - u10i * m0 + u11r * r1 - u11i * m1;
                si[i1] = u10r * m0 + u10i * r0 + u11r * m1 + u11i * r1;
            }
        }
        // CNOT(0,1): for b0=1 swap b1 -> swap idx 8..11 <-> 12..15
#pragma unroll
        for (int j = 0; j < 4; ++j) {
            float tr = sr[8 + j], ti = si[8 + j];
            sr[8 + j] = sr[12 + j]; si[8 + j] = si[12 + j];
            sr[12 + j] = tr;        si[12 + j] = ti;
        }
    }

    float P[4] = {0.f, 0.f, 0.f, 0.f};              // <Z_q>
#pragma unroll
    for (int idx = 0; idx < 16; ++idx) {
        float p = sr[idx] * sr[idx] + si[idx] * si[idx];
#pragma unroll
        for (int q = 0; q < 4; ++q)
            P[q] += ((idx >> (3 - q)) & 1) ? -p : p;
    }

    // ---- phase 1: h stripe -> LDS.  h[m][k] = relu(P[m] . W1[k,:]) ---------
    // thread (g = tid>>5, m = tid&31) fills chunks kc = g + 16*it.
    const int g16 = tid >> 5;                       // 0..15
#pragma unroll 2
    for (int it = 0; it < 16; ++it) {
        int kc = g16 + (it << 4);
        int kbase = kc << 3;
        v8h pack;
#pragma unroll
        for (int j = 0; j < 8; ++j) {
            float4 w = *(const float4*)(W1 + (size_t)(kbase + j) * 4);
            float h = fmaf(P[0], w.x, fmaf(P[1], w.y, fmaf(P[2], w.z, P[3] * w.w)));
            pack[j] = (_Float16)fmaxf(h, 0.f);
        }
        Hc[kc * 32 + m] = pack;                     // ds_store_b128, no conflict
    }
    __syncthreads();

    // ---- phase 2: wmma_f32_16x16x32_f16, double-buffered fragments ---------
    const int wave = tid >> 5;                      // 0..15
    const int lane = tid & 31;
    const int half = lane >> 4;                     // K-split lane group
    const int nrow = lane & 15;                     // A row / B column in tile

    const v8f zero = {0.f,0.f,0.f,0.f,0.f,0.f,0.f,0.f};
    v8f acc00 = zero, acc01 = zero, acc10 = zero, acc11 = zero;

    // B row pointers for N-tiles wave and wave+16 (lane's column nrow)
    const _Float16* bp0 = W2h + (size_t)((wave << 4) + nrow) * FF_DIM;
    const _Float16* bp1 = W2h + (size_t)(((wave + 16) << 4) + nrow) * FF_DIM;

    // A fragment (16-bit A 16x32, ISA 7.12.2): half h reads chunks
    // ks*4+h and ks*4+2+h of its token row.
    // B fragment (16-bit B 32x16): lane n, half h holds K[32ks+8h .. +7] and K[32ks+16+8h .. +7].

    // prologue: ks = 0
    v16h a0 = frag16(Hc[(half)*32 + nrow],          Hc[(2 + half)*32 + nrow]);
    v16h a1 = frag16(Hc[(half)*32 + 16 + nrow],     Hc[(2 + half)*32 + 16 + nrow]);
    int kb0 = (half << 3);
    v16h b0 = frag16(*(const v8h*)(bp0 + kb0), *(const v8h*)(bp0 + kb0 + 16));
    v16h b1 = frag16(*(const v8h*)(bp1 + kb0), *(const v8h*)(bp1 + kb0 + 16));

#pragma unroll 2
    for (int ks = 0; ks < 63; ++ks) {
        const int kn = ks + 1;
        // issue next-step fragment loads BEFORE this step's wmmas
        v16h a0n = frag16(Hc[(kn*4 + half)*32 + nrow],
                          Hc[(kn*4 + 2 + half)*32 + nrow]);
        v16h a1n = frag16(Hc[(kn*4 + half)*32 + 16 + nrow],
                          Hc[(kn*4 + 2 + half)*32 + 16 + nrow]);
        const int kb = (kn << 5) + (half << 3);
        v16h b0n = frag16(*(const v8h*)(bp0 + kb), *(const v8h*)(bp0 + kb + 16));
        v16h b1n = frag16(*(const v8h*)(bp1 + kb), *(const v8h*)(bp1 + kb + 16));

        acc00 = wmma_f16(a0, b0, acc00);
        acc01 = wmma_f16(a0, b1, acc01);
        acc10 = wmma_f16(a1, b0, acc10);
        acc11 = wmma_f16(a1, b1, acc11);

        a0 = a0n; a1 = a1n; b0 = b0n; b1 = b1n;   // dissolved by unroll 2
    }
    // tail: ks = 63
    acc00 = wmma_f16(a0, b0, acc00);
    acc01 = wmma_f16(a0, b1, acc01);
    acc10 = wmma_f16(a1, b0, acc10);
    acc11 = wmma_f16(a1, b1, acc11);

    // ---- epilogue: D layout (16x16 f32): VGPR i, lane L -> M = i + 8*half,
    // N = L&15.  out[tok, e].
#pragma unroll
    for (int i = 0; i < 8; ++i) {
        size_t r0 = (size_t)(blockIdx.x * TOKS_PER_BLOCK + i + half * 8) * E_DIM;
        size_t r1 = r0 + (size_t)16 * E_DIM;        // second M-tile
        out[r0 + (wave << 4) + nrow]        = acc00[i];
        out[r0 + ((wave + 16) << 4) + nrow] = acc01[i];
        out[r1 + (wave << 4) + nrow]        = acc10[i];
        out[r1 + ((wave + 16) << 4) + nrow] = acc11[i];
    }
}

// ---------------------------------------------------------------------------
extern "C" void kernel_launch(void* const* d_in, const int* in_sizes, int n_in,
                              void* d_out, int out_size, void* d_ws, size_t ws_size,
                              hipStream_t stream) {
    const float* x      = (const float*)d_in[0];   // [8,8192,512]
    const float* params = (const float*)d_in[1];   // [2,4,3]
    const float* W1     = (const float*)d_in[2];   // [2048,4]
    const float* W2     = (const float*)d_in[3];   // [512,2048]
    float* out = (float*)d_out;

    // workspace: [0,256) gates f32; [256, 256+2MB) W2 in f16
    float*    gates = (float*)d_ws;
    _Float16* W2h   = (_Float16*)((char*)d_ws + 256);

    qff_prep_gates<<<1, 32, 0, stream>>>(params, gates);
    qff_cvt_w2<<<(E_DIM * FF_DIM / 4) / 256, 256, 0, stream>>>(W2, (v4h*)W2h);
    qff_main<<<N_TOK / TOKS_PER_BLOCK, THREADS,
               /*dynamic LDS*/ 256 * 32 * 16, stream>>>(
        x, W1, W2h, gates, out);
}
